// SuperPoint_26182120636644
// MI455X (gfx1250) — matrix operand order repaired and checked by reading, not compiled, not run on hardware
//
#include <hip/hip_runtime.h>
#include <hip/hip_bf16.h>
#include <math.h>
#include <stdint.h>

typedef __attribute__((ext_vector_type(16))) _Float16 v16h;
typedef __attribute__((ext_vector_type(8)))  _Float16 v8h;
typedef __attribute__((ext_vector_type(8)))  float    v8f;

#define NMS_R     4
#define THRESH    0.005f
#define BORDER_PX 4
#define MAX_KPTS  1024

// ---------------------------------------------------------------------------
// Layer 1a: Cin=1 (K=9), too thin for WMMA -> direct conv, relu.
// ---------------------------------------------------------------------------
__global__ void conv1a_kernel(const float* __restrict__ img,
                              const float* __restrict__ w,
                              const float* __restrict__ bias,
                              float* __restrict__ out,
                              int H, int W, int total) {
  int i = blockIdx.x * blockDim.x + threadIdx.x;
  if (i >= total) return;
  int HW = H * W;
  int x = i % W;
  int y = (i / W) % H;
  int co = (i / HW) % 64;
  int b = i / (HW * 64);
  const float* im = img + (size_t)b * HW;
  float s = bias[co];
  for (int ky = 0; ky < 3; ky++) {
    int gy = y + ky - 1; if (gy < 0 || gy >= H) continue;
    for (int kx = 0; kx < 3; kx++) {
      int gx = x + kx - 1; if (gx < 0 || gx >= W) continue;
      s += w[co * 9 + ky * 3 + kx] * im[gy * W + gx];
    }
  }
  out[i] = fmaxf(s, 0.f);
}

// ---------------------------------------------------------------------------
// A-fragment build: 16x32 f16 per CDNA5 layout. Lane L: M=L%16;
// VGPR0-3 hold K=8*kh..8*kh+7 (2 halves each), VGPR4-7 hold K=8*kh+16..+23.
// With channel-innermost LDS these are two contiguous 16B reads.
// ---------------------------------------------------------------------------
__device__ inline v16h load_afrag(const _Float16* wrow, int koff) {
  v8h lo = *reinterpret_cast<const v8h*>(wrow + koff);
  v8h hi = *reinterpret_cast<const v8h*>(wrow + koff + 16);
  return __builtin_shufflevector(lo, hi, 0, 1, 2, 3, 4, 5, 6, 7,
                                 8, 9, 10, 11, 12, 13, 14, 15);
}

// ---------------------------------------------------------------------------
// Implicit-GEMM 3x3 conv (pad=1) via v_wmma_f32_16x16x32_f16.
// Block = 128 threads (4 waves). Block tile: 32 couts x 64 pixels (one row).
// Wave w handles pixels [x0+16w, x0+16w+15] and 2 cout sub-tiles (2 acc).
// LDS channel-innermost so B fragments are two contiguous 16B reads.
// ---------------------------------------------------------------------------
__global__ __launch_bounds__(128) void conv3x3_wmma(
    const float* __restrict__ in, const float* __restrict__ w,
    const float* __restrict__ bias, float* __restrict__ out,
    int Cin, int Cout, int H, int W, int relu, int nxseg) {
  __shared__ alignas(32) _Float16 inS[3 * 66 * 32];  // [row][col][c]
  __shared__ alignas(32) _Float16 wS[9 * 32 * 32];   // [kp][m(0..31)][c]
  const int b = blockIdx.z;
  const int HW = H * W;
  in  += (size_t)b * Cin * HW;
  out += (size_t)b * Cout * HW;
  const int y  = blockIdx.x / nxseg;
  const int x0 = (blockIdx.x % nxseg) * 64;
  const int coutBase = blockIdx.y * 32;
  const int tid  = threadIdx.x;
  const int lane = tid & 31;
  const int wave = tid >> 5;
  const int nIdx = lane & 15;     // A-row M / B-col N / D-col N
  const int kh   = lane >> 4;     // lane half selects K sub-range
  const int koff = kh * 8;
  const int kb   = kh * 16;

  v8f acc0 = {0.f, 0.f, 0.f, 0.f, 0.f, 0.f, 0.f, 0.f};
  v8f acc1 = {0.f, 0.f, 0.f, 0.f, 0.f, 0.f, 0.f, 0.f};

  for (int cb = 0; cb < Cin; cb += 32) {
    __syncthreads();
    // stage input tile (zero-padded halo), fp32 -> fp16, coalesced over cols
    for (int t = tid; t < 3 * 66 * 32; t += 128) {
      int pos = t % (3 * 66);
      int cc  = t / (3 * 66);
      int col = pos % 66;
      int rr  = pos / 66;
      int gy = y + rr - 1, gx = x0 + col - 1;
      float v = 0.f;
      if (gy >= 0 && gy < H && gx >= 0 && gx < W)
        v = in[(size_t)(cb + cc) * HW + gy * W + gx];
      inS[(rr * 66 + col) * 32 + cc] = (_Float16)v;
    }
    // stage 32x32 weight slice for all 9 taps
    for (int t = tid; t < 9 * 32 * 32; t += 128) {
      int c  = t % 32;
      int m  = (t / 32) % 32;
      int kp = t / (32 * 32);
      wS[t] = (_Float16)w[((size_t)(coutBase + m) * Cin + (cb + c)) * 9 + kp];
    }
    __syncthreads();

#pragma unroll
    for (int kp = 0; kp < 9; kp++) {
      const int ky = kp / 3, kx = kp % 3;
      // B fragment: 32x16 f16. Lane L: N=L%16; same K order as A:
      // K=8*kh..8*kh+7 then K=8*kh+16..+23 -> two 16B reads.
      const int col = wave * 16 + nIdx + kx;   // stage origin is x0-1
      v16h bf = load_afrag(&inS[(ky * 66 + col) * 32], koff);
      v16h a0 = load_afrag(&wS[(kp * 32 + nIdx) * 32], koff);
      v16h a1 = load_afrag(&wS[(kp * 32 + 16 + nIdx) * 32], koff);
      acc0 = __builtin_amdgcn_wmma_f32_16x16x32_f16(
          false, a0, false, bf, (short)0, acc0, false, false);
      acc1 = __builtin_amdgcn_wmma_f32_16x16x32_f16(
          false, a1, false, bf, (short)0, acc1, false, false);
    }
  }

  // Epilogue: D 16x16 f32: VGPR v -> M = v + 8*(lane>=16); N = lane%16.
  const int mb = kh * 8;
  const int xg = x0 + wave * 16 + nIdx;
  if (xg < W) {
#pragma unroll
    for (int v = 0; v < 8; v++) {
      int co = coutBase + mb + v;
      float r = acc0[v] + bias[co];
      if (relu) r = fmaxf(r, 0.f);
      out[(size_t)co * HW + y * W + xg] = r;
    }
#pragma unroll
    for (int v = 0; v < 8; v++) {
      int co = coutBase + 16 + mb + v;
      float r = acc1[v] + bias[co];
      if (relu) r = fmaxf(r, 0.f);
      out[(size_t)co * HW + y * W + xg] = r;
    }
  }
}

// ---------------------------------------------------------------------------
// 1x1 conv GEMM via WMMA: Cout x (H*W) = W[Cout,Cin] * X[Cin,HW].
// Block tile: 32 couts x 64 flat pixels. Handles Cout not multiple of 32.
// ---------------------------------------------------------------------------
__global__ __launch_bounds__(128) void conv1x1_wmma(
    const float* __restrict__ in, const float* __restrict__ w,
    const float* __restrict__ bias, float* __restrict__ out,
    int Cin, int Cout, int HW, int relu) {
  __shared__ alignas(32) _Float16 inS[64 * 32];   // [px][c]
  __shared__ alignas(32) _Float16 wS[32 * 32];    // [m][c]
  const int b = blockIdx.z;
  in  += (size_t)b * Cin * HW;
  out += (size_t)b * Cout * HW;
  const int p0 = blockIdx.x * 64;
  const int coutBase = blockIdx.y * 32;
  const int tid  = threadIdx.x;
  const int lane = tid & 31;
  const int wave = tid >> 5;
  const int nIdx = lane & 15;
  const int kh   = lane >> 4;
  const int koff = kh * 8;
  const int kb   = kh * 16;

  v8f acc0 = {0.f, 0.f, 0.f, 0.f, 0.f, 0.f, 0.f, 0.f};
  v8f acc1 = {0.f, 0.f, 0.f, 0.f, 0.f, 0.f, 0.f, 0.f};

  for (int cb = 0; cb < Cin; cb += 32) {
    __syncthreads();
    for (int t = tid; t < 64 * 32; t += 128) {
      int px = t % 64;
      int cc = t / 64;
      float v = (p0 + px < HW) ? in[(size_t)(cb + cc) * HW + p0 + px] : 0.f;
      inS[px * 32 + cc] = (_Float16)v;
    }
    for (int t = tid; t < 32 * 32; t += 128) {
      int c = t % 32;
      int m = t / 32;
      int co = coutBase + m;
      float v = (co < Cout) ? w[(size_t)co * Cin + cb + c] : 0.f;
      wS[t] = (_Float16)v;
    }
    __syncthreads();

    const int pxl = wave * 16 + nIdx;
    v16h bf = load_afrag(&inS[pxl * 32], koff);
    v16h a0 = load_afrag(&wS[nIdx * 32], koff);
    v16h a1 = load_afrag(&wS[(16 + nIdx) * 32], koff);
    acc0 = __builtin_amdgcn_wmma_f32_16x16x32_f16(
        false, a0, false, bf, (short)0, acc0, false, false);
    acc1 = __builtin_amdgcn_wmma_f32_16x16x32_f16(
        false, a1, false, bf, (short)0, acc1, false, false);
  }

  const int mb = kh * 8;
  const int p  = p0 + wave * 16 + nIdx;
  if (p < HW) {
#pragma unroll
    for (int v = 0; v < 8; v++) {
      int co = coutBase + mb + v;
      if (co < Cout) {
        float r = acc0[v] + bias[co];
        if (relu) r = fmaxf(r, 0.f);
        out[(size_t)co * HW + p] = r;
      }
    }
#pragma unroll
    for (int v = 0; v < 8; v++) {
      int co = coutBase + 16 + mb + v;
      if (co < Cout) {
        float r = acc1[v] + bias[co];
        if (relu) r = fmaxf(r, 0.f);
        out[(size_t)co * HW + p] = r;
      }
    }
  }
}

// ---------------------------------------------------------------------------
// 2x2 max pool, stride 2.
// ---------------------------------------------------------------------------
__global__ void pool2_kernel(const float* __restrict__ in, float* __restrict__ out,
                             int C, int H, int W, int total) {
  int i = blockIdx.x * blockDim.x + threadIdx.x;
  if (i >= total) return;
  int Ho = H / 2, Wo = W / 2;
  int x = i % Wo;
  int y = (i / Wo) % Ho;
  int c = (i / (Wo * Ho)) % C;
  int b = i / (Wo * Ho * C);
  const float* p = in + ((size_t)b * C + c) * H * W + (2 * y) * W + 2 * x;
  out[i] = fmaxf(fmaxf(p[0], p[1]), fmaxf(p[W], p[W + 1]));
}

// ---------------------------------------------------------------------------
// Detector head: softmax over 65 ch, drop dustbin, 8x8 depth-to-space.
// ---------------------------------------------------------------------------
__global__ void softmax_d2s_kernel(const float* __restrict__ raw,
                                   float* __restrict__ scores,
                                   int h, int w, int total) {
  int i = blockIdx.x * blockDim.x + threadIdx.x;
  if (i >= total) return;
  int hw = h * w;
  int b = i / hw, p = i % hw;
  int y = p / w, x = p % w;
  const float* r = raw + (size_t)b * 65 * hw + p;
  float mx = r[0];
  for (int c = 1; c < 65; c++) mx = fmaxf(mx, r[(size_t)c * hw]);
  float s = 0.f;
  for (int c = 0; c < 65; c++) s += expf(r[(size_t)c * hw] - mx);
  float inv = 1.f / s;
  float* so = scores + (size_t)b * (h * 8) * (w * 8);
  for (int c = 0; c < 64; c++) {
    float v = expf(r[(size_t)c * hw] - mx) * inv;
    int ry = c / 8, rx = c % 8;
    so[(y * 8 + ry) * (w * 8) + (x * 8 + rx)] = v;
  }
}

// ---------------------------------------------------------------------------
// Descriptor map L2-normalize over 256 channels (in place).
// ---------------------------------------------------------------------------
__global__ void dnorm_kernel(float* __restrict__ dmap, int C, int hw, int total) {
  int i = blockIdx.x * blockDim.x + threadIdx.x;
  if (i >= total) return;
  int b = i / hw, p = i % hw;
  float* d = dmap + (size_t)b * C * hw + p;
  float s = 0.f;
  for (int c = 0; c < C; c++) { float v = d[(size_t)c * hw]; s += v * v; }
  float inv = 1.f / fmaxf(sqrtf(s), 1e-12f);
  for (int c = 0; c < C; c++) d[(size_t)c * hw] *= inv;
}

// ---------------------------------------------------------------------------
// 9x9 SAME max-pool helper (window always contains center -> init with it).
// ---------------------------------------------------------------------------
__device__ inline float pool9_at(const float* s, int y, int x, int H, int W) {
  float m = s[y * W + x];
  for (int dy = -NMS_R; dy <= NMS_R; dy++) {
    int yy = y + dy; if (yy < 0 || yy >= H) continue;
    for (int dx = -NMS_R; dx <= NMS_R; dx++) {
      int xx = x + dx; if (xx < 0 || xx >= W) continue;
      m = fmaxf(m, s[yy * W + xx]);
    }
  }
  return m;
}

__global__ void nms_init_kernel(const float* __restrict__ scores,
                                float* __restrict__ mask, int H, int W, int total) {
  int i = blockIdx.x * blockDim.x + threadIdx.x;
  if (i >= total) return;
  int b = i / (H * W), p = i % (H * W);
  const float* s = scores + (size_t)b * H * W;
  mask[i] = (s[p] == pool9_at(s, p / W, p % W, H, W)) ? 1.f : 0.f;
}

__global__ void nms_pool_kernel(const float* __restrict__ src,
                                float* __restrict__ dst, int H, int W, int total) {
  int i = blockIdx.x * blockDim.x + threadIdx.x;
  if (i >= total) return;
  int b = i / (H * W), p = i % (H * W);
  const float* s = src + (size_t)b * H * W;
  dst[i] = pool9_at(s, p / W, p % W, H, W);
}

__global__ void nms_supp_kernel(const float* __restrict__ scores,
                                const float* __restrict__ pooled_mask,
                                float* __restrict__ supp_scores, int total) {
  int i = blockIdx.x * blockDim.x + threadIdx.x;
  if (i >= total) return;
  supp_scores[i] = (pooled_mask[i] > 0.f) ? 0.f : scores[i];
}

__global__ void nms_update_kernel(const float* __restrict__ supp_scores,
                                  const float* __restrict__ pooled_mask,
                                  float* __restrict__ mask, int H, int W, int total) {
  int i = blockIdx.x * blockDim.x + threadIdx.x;
  if (i >= total) return;
  int b = i / (H * W), p = i % (H * W);
  const float* s2 = supp_scores + (size_t)b * H * W;
  float pl = pool9_at(s2, p / W, p % W, H, W);
  bool supp = pooled_mask[i] > 0.f;
  bool nm = (s2[p] == pl) && !supp;
  mask[i] = (mask[i] > 0.f || nm) ? 1.f : 0.f;
}

__global__ void nms_masked_kernel(const float* __restrict__ scores,
                                  const float* __restrict__ mask,
                                  float* __restrict__ masked,
                                  int H, int W, int total) {
  int i = blockIdx.x * blockDim.x + threadIdx.x;
  if (i >= total) return;
  int p = i % (H * W);
  int y = p / W, x = p % W;
  bool border = (y >= BORDER_PX && y < H - BORDER_PX &&
                 x >= BORDER_PX && x < W - BORDER_PX);
  masked[i] = (mask[i] > 0.f && scores[i] > THRESH && border) ? scores[i] : -1.f;
}

// ---------------------------------------------------------------------------
// Per-batch iterative top-1024 (block argmax, lowest-index tie-break).
// Destructive on `masked` (recomputed each launch -> deterministic).
// ---------------------------------------------------------------------------
__global__ void topk_kernel(float* __restrict__ masked,
                            float* __restrict__ out_kpts,
                            float* __restrict__ out_scores,
                            int H, int W) {
  const int b = blockIdx.x;
  float* m = masked + (size_t)b * H * W;
  __shared__ float sv[256];
  __shared__ int   si[256];
  const int tid = threadIdx.x;
  const int N = H * W;
  for (int k = 0; k < MAX_KPTS; k++) {
    float best = -3.0e38f;
    int bi = 0x7fffffff;
    for (int i = tid; i < N; i += 256) {
      float v = m[i];
      if (v > best || (v == best && i < bi)) { best = v; bi = i; }
    }
    sv[tid] = best; si[tid] = bi;
    __syncthreads();
    for (int s = 128; s > 0; s >>= 1) {
      if (tid < s) {
        if (sv[tid + s] > sv[tid] ||
            (sv[tid + s] == sv[tid] && si[tid + s] < si[tid])) {
          sv[tid] = sv[tid + s]; si[tid] = si[tid + s];
        }
      }
      __syncthreads();
    }
    if (tid == 0) {
      float v = sv[0];
      int idx = si[0];
      bool valid = v > 0.f;
      out_kpts[((size_t)b * MAX_KPTS + k) * 2 + 0] = valid ? (float)(idx % W) : 0.f;
      out_kpts[((size_t)b * MAX_KPTS + k) * 2 + 1] = valid ? (float)(idx / W) : 0.f;
      out_scores[(size_t)b * MAX_KPTS + k] = valid ? v : 0.f;
      m[idx] = -2.f;
    }
    __syncthreads();
  }
}

// ---------------------------------------------------------------------------
// Bilinear descriptor sampling + L2 norm over 256 channels. 1 block per kpt,
// 256 threads = 256 channels.
// ---------------------------------------------------------------------------
__global__ void sample_desc_kernel(const float* __restrict__ dmap,
                                   const float* __restrict__ out_kpts,
                                   const float* __restrict__ out_scores,
                                   float* __restrict__ out_desc,
                                   int h, int w) {
  const int bk = blockIdx.x;
  const int b = bk / MAX_KPTS;
  const int k = bk % MAX_KPTS;
  const int c = threadIdx.x;
  float x = out_kpts[((size_t)b * MAX_KPTS + k) * 2 + 0];
  float y = out_kpts[((size_t)b * MAX_KPTS + k) * 2 + 1];
  bool valid = out_scores[(size_t)b * MAX_KPTS + k] > 0.f;
  float gx = x / (w * 8 * 0.5f) * 2.f - 1.f;
  float gy = y / (h * 8 * 0.5f) * 2.f - 1.f;
  float px = (gx + 1.f) * 0.5f * (float)(w - 1);
  float py = (gy + 1.f) * 0.5f * (float)(h - 1);
  int x0 = (int)floorf(px), y0 = (int)floorf(py);
  float wx1 = px - x0, wx0 = 1.f - wx1;
  float wy1 = py - y0, wy0 = 1.f - wy1;
  const float* dm = dmap + ((size_t)b * 256 + c) * h * w;

  float v = 0.f;
  {
    int ys[2] = {y0, y0 + 1};
    int xs[2] = {x0, x0 + 1};
    float wy[2] = {wy0, wy1};
    float wx[2] = {wx0, wx1};
    for (int iy = 0; iy < 2; iy++)
      for (int ix = 0; ix < 2; ix++) {
        int yi = ys[iy], xi = xs[ix];
        float ok = (xi >= 0 && xi < w && yi >= 0 && yi < h) ? 1.f : 0.f;
        int yc = min(max(yi, 0), h - 1);
        int xc = min(max(xi, 0), w - 1);
        v += dm[yc * w + xc] * ok * wy[iy] * wx[ix];
      }
  }
  __shared__ float red[256];
  red[c] = v * v;
  __syncthreads();
  for (int s = 128; s > 0; s >>= 1) {
    if (c < s) red[c] += red[c + s];
    __syncthreads();
  }
  float nrm = fmaxf(sqrtf(red[0]), 1e-12f);
  out_desc[((size_t)b * 256 + c) * MAX_KPTS + k] = valid ? v / nrm : 0.f;
}

// ---------------------------------------------------------------------------
// Host-side launcher
// ---------------------------------------------------------------------------
extern "C" void kernel_launch(void* const* d_in, const int* in_sizes, int n_in,
                              void* d_out, int out_size, void* d_ws, size_t ws_size,
                              hipStream_t stream) {
  const int H0 = 480, W0 = 640;
  const int B = in_sizes[0] / (H0 * W0);

  const float* image = (const float*)d_in[0];
  const float* w1a = (const float*)d_in[1];  const float* b1a = (const float*)d_in[2];
  const float* w1b = (const float*)d_in[3];  const float* b1b = (const float*)d_in[4];
  const float* w2a = (const float*)d_in[5];  const float* b2a = (const float*)d_in[6];
  const float* w2b = (const float*)d_in[7];  const float* b2b = (const float*)d_in[8];
  const float* w3a = (const float*)d_in[9];  const float* b3a = (const float*)d_in[10];
  const float* w3b = (const float*)d_in[11]; const float* b3b = (const float*)d_in[12];
  const float* w4a = (const float*)d_in[13]; const float* b4a = (const float*)d_in[14];
  const float* w4b = (const float*)d_in[15]; const float* b4b = (const float*)d_in[16];
  const float* wpa = (const float*)d_in[17]; const float* bpa = (const float*)d_in[18];
  const float* wpb = (const float*)d_in[19]; const float* bpb = (const float*)d_in[20];
  const float* wda = (const float*)d_in[21]; const float* bda = (const float*)d_in[22];
  const float* wdb = (const float*)d_in[23]; const float* bdb = (const float*)d_in[24];

  // workspace carve
  uint8_t* wp = (uint8_t*)d_ws;
  auto carve = [&](size_t nfloats) -> float* {
    float* r = (float*)wp;
    wp += ((nfloats * sizeof(float)) + 255) & ~(size_t)255;
    return r;
  };
  float* actA  = carve((size_t)B * 64 * H0 * W0);     // ping
  float* actB  = carve((size_t)B * 64 * H0 * W0);     // pong
  float* feat  = carve((size_t)B * 128 * 60 * 80);    // conv4b output
  float* headA = carve((size_t)B * 256 * 60 * 80);    // cPa / cDa
  float* buf65 = carve((size_t)B * 65 * 60 * 80);     // detector logits
  float* dmap  = carve((size_t)B * 256 * 60 * 80);    // descriptor map
  float* scoresF = carve((size_t)B * H0 * W0);        // full-res scores
  float* maskB   = carve((size_t)B * H0 * W0);
  float* tmp1    = carve((size_t)B * H0 * W0);        // pooled mask / masked
  float* tmp2    = carve((size_t)B * H0 * W0);        // suppressed scores

  float* out_kpts   = (float*)d_out;                      // (B, K, 2)
  float* out_scores = out_kpts + (size_t)B * MAX_KPTS * 2;
  float* out_desc   = out_scores + (size_t)B * MAX_KPTS;  // (B, 256, K)

  auto c3 = [&](const float* in, const float* w, const float* bias, float* out,
                int Cin, int Cout, int H, int W, int relu) {
    int nxseg = (W + 63) / 64;
    dim3 g(nxseg * H, Cout / 32, B);
    conv3x3_wmma<<<g, 128, 0, stream>>>(in, w, bias, out, Cin, Cout, H, W, relu, nxseg);
  };
  auto c1 = [&](const float* in, const float* w, const float* bias, float* out,
                int Cin, int Cout, int HW, int relu) {
    dim3 g((HW + 63) / 64, (Cout + 31) / 32, B);
    conv1x1_wmma<<<g, 128, 0, stream>>>(in, w, bias, out, Cin, Cout, HW, relu);
  };
  auto pool = [&](const float* in, float* out, int C, int H, int W) {
    int total = B * C * (H / 2) * (W / 2);
    pool2_kernel<<<(total + 255) / 256, 256, 0, stream>>>(in, out, C, H, W, total);
  };

  // ---- backbone ----
  {
    int total = B * 64 * H0 * W0;
    conv1a_kernel<<<(total + 255) / 256, 256, 0, stream>>>(image, w1a, b1a, actA, H0, W0, total);
  }
  c3(actA, w1b, b1b, actB, 64, 64, 480, 640, 1);
  pool(actB, actA, 64, 480, 640);                       // -> 240x320
  c3(actA, w2a, b2a, actB, 64, 64, 240, 320, 1);
  c3(actB, w2b, b2b, actA, 64, 64, 240, 320, 1);
  pool(actA, actB, 64, 240, 320);                       // -> 120x160
  c3(actB, w3a, b3a, actA, 64, 128, 120, 160, 1);
  c3(actA, w3b, b3b, actB, 128, 128, 120, 160, 1);
  pool(actB, actA, 128, 120, 160);                      // -> 60x80
  c3(actA, w4a, b4a, actB, 128, 128, 60, 80, 1);
  c3(actB, w4b, b4b, feat, 128, 128, 60, 80, 1);

  // ---- detector head ----
  c3(feat, wpa, bpa, headA, 128, 256, 60, 80, 1);
  c1(headA, wpb, bpb, buf65, 256, 65, 60 * 80, 0);
  {
    int total = B * 60 * 80;
    softmax_d2s_kernel<<<(total + 255) / 256, 256, 0, stream>>>(buf65, scoresF, 60, 80, total);
  }

  // ---- descriptor head ----
  c3(feat, wda, bda, headA, 128, 256, 60, 80, 1);
  c1(headA, wdb, bdb, dmap, 256, 256, 60 * 80, 0);
  {
    int total = B * 60 * 80;
    dnorm_kernel<<<(total + 255) / 256, 256, 0, stream>>>(dmap, 256, 60 * 80, total);
  }

  // ---- iterative NMS ----
  {
    int total = B * H0 * W0;
    int nb = (total + 255) / 256;
    nms_init_kernel<<<nb, 256, 0, stream>>>(scoresF, maskB, H0, W0, total);
    for (int it = 0; it < 2; it++) {
      nms_pool_kernel<<<nb, 256, 0, stream>>>(maskB, tmp1, H0, W0, total);
      nms_supp_kernel<<<nb, 256, 0, stream>>>(scoresF, tmp1, tmp2, total);
      nms_update_kernel<<<nb, 256, 0, stream>>>(tmp2, tmp1, maskB, H0, W0, total);
    }
    nms_masked_kernel<<<nb, 256, 0, stream>>>(scoresF, maskB, tmp1, H0, W0, total);
  }

  // ---- top-k keypoints ----
  topk_kernel<<<B, 256, 0, stream>>>(tmp1, out_kpts, out_scores, H0, W0);

  // ---- descriptor sampling ----
  sample_desc_kernel<<<B * MAX_KPTS, 256, 0, stream>>>(dmap, out_kpts, out_scores,
                                                       out_desc, 60, 80);
}
